// TransformerAttention_31009663877344
// MI455X (gfx1250) — matrix operand order repaired and checked by reading, not compiled, not run on hardware
//
#include <hip/hip_runtime.h>
#include <math.h>
#include <stdint.h>

// ---------------------------------------------------------------------------
// Problem constants (match reference)
// ---------------------------------------------------------------------------
static constexpr int Bc   = 8;
static constexpr int Sc   = 512;
static constexpr int Dc   = 512;
static constexpr int DFFc = 2048;
static constexpr int Hc   = 8;
static constexpr int Nc   = 6;
static constexpr int DKc  = 64;     // Dc / Hc

typedef __attribute__((ext_vector_type(16))) _Float16 v16h;
typedef __attribute__((ext_vector_type(8)))  _Float16 v8h;
typedef __attribute__((ext_vector_type(8)))  float    v8f;
typedef __attribute__((ext_vector_type(4)))  unsigned int u32x4;
typedef __attribute__((ext_vector_type(8)))  int          i32x8;
typedef __attribute__((ext_vector_type(4)))  int          i32x4;

// Tensor Data Mover availability (compile-safe fallback to manual copies)
#if defined(__HIP_DEVICE_COMPILE__) && __has_builtin(__builtin_amdgcn_tensor_load_to_lds)
#define HAS_TDM 1
#else
#define HAS_TDM 0
#endif

// ---------------------------------------------------------------------------
// TDM tile load: 2D tile of 16-bit data, tile_dim0 = 32 elems (64B rows),
// tile_dim1 = 128 rows, global row stride = ld elements, LDS pitch padded
// 32 halves + 8 halves (pad_interval = 16 DWORDs, pad_amount = 4 DWORDs).
// Rows >= valid_rows are zero-filled by TDM OOB handling (tensor_dim1).
// ---------------------------------------------------------------------------
__device__ __forceinline__
void tdm_load_tile(const _Float16* gsrc, unsigned lds_byte_addr,
                   int ld_elems, int valid_rows)
{
#if HAS_TDM
    const unsigned long long ga = (unsigned long long)(uintptr_t)gsrc;
    u32x4 g0;
    g0[0] = 1u;                                         // count=1, user D#
    g0[1] = lds_byte_addr;                              // lds_addr
    g0[2] = (unsigned)(ga & 0xFFFFFFFFull);             // global_addr lo
    g0[3] = (unsigned)((ga >> 32) & 0x01FFFFFFull)      // global_addr hi
          | (2u << 30);                                 // type = 2 (image)
    i32x8 g1;
    g1[0] = (1 << 16)        // data_size = 1 -> 2 bytes
          | (1 << 20)        // pad_enable
          | (3 << 22)        // pad_interval: 16 DWORDs (64B) before pad
          | (3 << 25);       // pad_amount: 4 DWORDs (16B = 8 halves)
    g1[1] = (32 << 16);                                 // tensor_dim0 lo16 = 32
    g1[2] = ((unsigned)(valid_rows & 0xFFFF)) << 16;    // tensor_dim1 lo16
    g1[3] = (32 << 16);                                 // tile_dim0 = 32
    g1[4] = 128;                                        // tile_dim1 = 128
    g1[5] = ld_elems;                                   // tensor_dim0_stride lo32
    g1[6] = 0;
    g1[7] = 0;
    const i32x4 z4 = {0, 0, 0, 0};
#if defined(__clang_major__) && (__clang_major__ >= 23)
    const i32x8 z8 = {0, 0, 0, 0, 0, 0, 0, 0};
    __builtin_amdgcn_tensor_load_to_lds(g0, g1, z4, z4, z8, 0);
#else
    __builtin_amdgcn_tensor_load_to_lds(g0, g1, z4, z4, 0);
#endif
#else
    (void)gsrc; (void)lds_byte_addr; (void)ld_elems; (void)valid_rows;
#endif
}

__device__ __forceinline__ void tdm_wait()
{
#if HAS_TDM
#if __has_builtin(__builtin_amdgcn_s_wait_tensorcnt)
    __builtin_amdgcn_s_wait_tensorcnt(0);
#else
    asm volatile("s_wait_tensorcnt 0x0" ::: "memory");
#endif
#endif
}

// ---------------------------------------------------------------------------
// All-f16-operand WMMA GEMM:
//   A: f16 [M,K] row-major (lda);  B: f16 [N,K] row-major (ldb)
//   C: OUTH ? f16 : f32, + bias (f32), optional residual (f32), optional ReLU
//   Batch z decomposed zo=z/inner, zi=z%inner with per-operand elem strides.
// Block 256 thr = 8 waves; tile 128x128, K-step 32; double-buffered LDS with
// TDM (wave 0 issues 2 descriptor loads / step) or manual 16B copies.
// ---------------------------------------------------------------------------
#define BMT 128
#define BNT 128
#define BKT 32
#define LDT 40   // LDS pitch in halves (80B rows; 16B aligned subchunks)

template<bool RELU, bool RES, bool OUTH>
__global__ __launch_bounds__(256)
void gemm_h(const _Float16* __restrict__ A, long sAo, long sAi,
            const _Float16* __restrict__ Bm, long sBo, long sBi,
            const float* __restrict__ bias,
            const float* __restrict__ Rr, long sRo, long sRi,
            void* __restrict__ Cv, long sCo, long sCi,
            int inner, int M, int Nn, int K, int lda, int ldb, int ldc)
{
    __shared__ _Float16 At[2][BMT * LDT];
    __shared__ _Float16 Bt[2][BNT * LDT];

    const int z  = blockIdx.z;
    const int zo = z / inner, zi = z % inner;
    A  += (long)zo * sAo + (long)zi * sAi;
    Bm += (long)zo * sBo + (long)zi * sBi;
    float*    Cf = nullptr;
    _Float16* Ch = nullptr;
    if (OUTH) Ch = (_Float16*)Cv + (long)zo * sCo + (long)zi * sCi;
    else      Cf = (float*)Cv    + (long)zo * sCo + (long)zi * sCi;
    if (RES)  Rr += (long)zo * sRo + (long)zi * sRi;

    const int m0   = blockIdx.x * BMT;
    const int n0   = blockIdx.y * BNT;
    const int t    = threadIdx.x;
    const int lane = t & 31;
    const int w    = t >> 5;
    const int wm   = w >> 2;          // 0..1 -> 64-row strip
    const int wn   = w & 3;           // 0..3 -> 32-col strip
    const int laneHi = lane >> 4;
    const int lr     = lane & 15;

    const _Float16* Ab = A  + (long)m0 * lda;
    const _Float16* Bb = Bm + (long)n0 * ldb;
    const int validB = (Nn - n0) < BNT ? (Nn - n0) : BNT;

    // pre-zero B rows beyond Nn (persist across iterations; staging skips them)
    if (validB < BNT) {
        for (int i = t; i < 2 * BNT * LDT; i += 256) (&Bt[0][0])[i] = (_Float16)0;
    }

    auto stage = [&](int k0, int buf) {
#if HAS_TDM
        if (w == 0) {
            tdm_load_tile(Ab + k0, (unsigned)(uintptr_t)&At[buf][0], lda, BMT);
            tdm_load_tile(Bb + k0, (unsigned)(uintptr_t)&Bt[buf][0], ldb, validB);
        }
#else
#pragma unroll
        for (int i = 0; i < 2; ++i) {
            const int q = t + i * 256;            // 0..511
            const int r = q >> 2;                 // 0..127
            const int c = (q & 3) * 8;            // 0,8,16,24
            *(float4*)&At[buf][r * LDT + c] =
                *(const float4*)(Ab + (long)r * lda + k0 + c);
            if (r < validB)
                *(float4*)&Bt[buf][r * LDT + c] =
                    *(const float4*)(Bb + (long)r * ldb + k0 + c);
        }
#endif
    };

    v8f acc[4][2] = {};

    stage(0, 0);
    if (w == 0) tdm_wait();
    __syncthreads();

    int cur = 0;
    for (int k0 = 0; k0 < K; k0 += BKT) {
        const int nxt = cur ^ 1;
        if (k0 + BKT < K) stage(k0 + BKT, nxt);   // overlap DMA with compute

        // ---- B fragments (K chunks at laneHi*8 and 16+laneHi*8)
        union { v16h v; v8h h[2]; } bf[2];
#pragma unroll
        for (int nt = 0; nt < 2; ++nt) {
            const _Float16* bp = &Bt[cur][(wn * 32 + nt * 16 + lr) * LDT + laneHi * 8];
            bf[nt].h[0] = *(const v8h*)(bp);
            bf[nt].h[1] = *(const v8h*)(bp + 16);
        }
        // ---- A fragments + 8 WMMA (K chunks at laneHi*8 and 16+laneHi*8)
#pragma unroll
        for (int mt = 0; mt < 4; ++mt) {
            union { v16h v; v8h h[2]; } af;
            const _Float16* ap = &At[cur][(wm * 64 + mt * 16 + lr) * LDT + laneHi * 8];
            af.h[0] = *(const v8h*)(ap);
            af.h[1] = *(const v8h*)(ap + 16);
#pragma unroll
            for (int nt = 0; nt < 2; ++nt) {
                acc[mt][nt] = __builtin_amdgcn_wmma_f32_16x16x32_f16(
                    false, af.v, false, bf[nt].v, (short)0, acc[mt][nt],
                    false, false);
            }
        }
        if (k0 + BKT < K && w == 0) tdm_wait();
        __syncthreads();
        cur = nxt;
    }

    // ---- epilogue: D layout (m = r + 8*laneHi, n = lane%16)
#pragma unroll
    for (int nt = 0; nt < 2; ++nt) {
        const int col = n0 + wn * 32 + nt * 16 + lr;
        if (col >= Nn) continue;
        const float bsv = bias ? bias[col] : 0.f;
#pragma unroll
        for (int mt = 0; mt < 4; ++mt) {
#pragma unroll
            for (int r = 0; r < 8; ++r) {
                const int row = m0 + wm * 64 + mt * 16 + r + 8 * laneHi;
                float v = acc[mt][nt][r] + bsv;
                if (RES)  v += Rr[(long)row * ldc + col];
                if (RELU) v = fmaxf(v, 0.f);
                if (OUTH) Ch[(long)row * ldc + col] = (_Float16)v;
                else      Cf[(long)row * ldc + col] = v;
            }
        }
    }
}

// ---------------------------------------------------------------------------
// Weight transpose-convert: fp32 [K,N] row-major -> f16 [N,K] row-major.
// grid (K/32, N/32), 256 threads, 32x32 LDS tile.
// ---------------------------------------------------------------------------
__global__ __launch_bounds__(256)
void tcvt_kernel(const float* __restrict__ wsrc, _Float16* __restrict__ wt,
                 int K, int Nn)
{
    __shared__ float tile[32][33];
    const int k0 = blockIdx.x * 32, n0 = blockIdx.y * 32;
    const int tx = threadIdx.x & 31, ty = threadIdx.x >> 5;   // ty 0..7
#pragma unroll
    for (int i = 0; i < 32; i += 8)
        tile[ty + i][tx] = wsrc[(size_t)(k0 + ty + i) * Nn + n0 + tx];
    __syncthreads();
#pragma unroll
    for (int i = 0; i < 32; i += 8)
        wt[(size_t)(n0 + ty + i) * K + k0 + tx] = (_Float16)tile[tx][ty + i];
}

// ---------------------------------------------------------------------------
// V transpose (f16->f16): Vh [B*S, D] head-sliced [S,DK] -> Vt [B*H][DK][S]
// ---------------------------------------------------------------------------
__global__ __launch_bounds__(256)
void vtrans_kernel(const _Float16* __restrict__ v, _Float16* __restrict__ vt)
{
    __shared__ _Float16 tile[32][33];
    const int z = blockIdx.z;                  // b*H + h
    const int b = z >> 3, h = z & 7;
    const _Float16* src = v + (size_t)b * Sc * Dc + h * DKc;   // [S,DK] stride D
    _Float16* dst = vt + (size_t)z * DKc * Sc;                 // [DK,S]
    const int s0 = blockIdx.x * 32, d0 = blockIdx.y * 32;
    const int tx = threadIdx.x & 31, ty = threadIdx.x >> 5;
#pragma unroll
    for (int i = 0; i < 32; i += 8)
        tile[ty + i][tx] = src[(size_t)(s0 + ty + i) * Dc + d0 + tx];
    __syncthreads();
#pragma unroll
    for (int i = 0; i < 32; i += 8)
        dst[(size_t)(d0 + ty + i) * Sc + s0 + tx] = tile[tx][ty + i];
}

// ---------------------------------------------------------------------------
// LayerNorm (torch-style: unbiased std, eps added to std), dual fp32+f16 out.
// ---------------------------------------------------------------------------
__global__ __launch_bounds__(256)
void layernorm_kernel(const float* __restrict__ x, const float* __restrict__ sc,
                      const float* __restrict__ bi, float* __restrict__ y,
                      _Float16* __restrict__ yh)
{
    const int row = blockIdx.x;
    const int t   = threadIdx.x;
    const float* xr = x + (long)row * Dc;
    const float v0 = xr[t], v1 = xr[t + 256];

    __shared__ float s1[256], s2[256];
    s1[t] = v0 + v1;
    s2[t] = v0 * v0 + v1 * v1;
    __syncthreads();
    for (int o = 128; o > 0; o >>= 1) {
        if (t < o) { s1[t] += s1[t + o]; s2[t] += s2[t + o]; }
        __syncthreads();
    }
    const float mean = s1[0] * (1.f / Dc);
    const float var  = fmaxf((s2[0] - (float)Dc * mean * mean) / (float)(Dc - 1), 0.f);
    const float inv  = 1.f / (sqrtf(var) + 1e-6f);
    const float o0 = sc[t]       * (v0 - mean) * inv + bi[t];
    const float o1 = sc[t + 256] * (v1 - mean) * inv + bi[t + 256];
    y[(long)row * Dc + t]        = o0;
    y[(long)row * Dc + t + 256]  = o1;
    yh[(long)row * Dc + t]       = (_Float16)o0;
    yh[(long)row * Dc + t + 256] = (_Float16)o1;
}

// ---------------------------------------------------------------------------
// Masked + scaled softmax, f16 in / f16 out, in place.  grid (S, B*H).
// ---------------------------------------------------------------------------
__global__ __launch_bounds__(256)
void softmax_kernel(_Float16* __restrict__ scores, const int* __restrict__ mask,
                    long mB, long mQ, float scale)
{
    const int q = blockIdx.x;
    const int z = blockIdx.y;         // b*H + h
    const int b = z / Hc;
    const int t = threadIdx.x;

    _Float16* row = scores + ((long)z * Sc + q) * Sc;
    const int* mrow = mask + (long)b * mB + (long)q * mQ;

    float x0 = (float)row[t] * scale;
    float x1 = (float)row[t + 256] * scale;
    if (mrow[t] == 0)       x0 = -1e9f;
    if (mrow[t + 256] == 0) x1 = -1e9f;

    __shared__ float sm[256];
    sm[t] = fmaxf(x0, x1);
    __syncthreads();
    for (int o = 128; o > 0; o >>= 1) {
        if (t < o) sm[t] = fmaxf(sm[t], sm[t + o]);
        __syncthreads();
    }
    const float mx = sm[0];
    __syncthreads();
    const float e0 = __expf(x0 - mx);
    const float e1 = __expf(x1 - mx);
    sm[t] = e0 + e1;
    __syncthreads();
    for (int o = 128; o > 0; o >>= 1) {
        if (t < o) sm[t] += sm[t + o];
        __syncthreads();
    }
    const float inv = 1.f / sm[0];
    row[t]       = (_Float16)(e0 * inv);
    row[t + 256] = (_Float16)(e1 * inv);
}

// ---------------------------------------------------------------------------
// Elementwise helpers
// ---------------------------------------------------------------------------
__global__ __launch_bounds__(256)
void add_pe_kernel(const float* __restrict__ x, const float* __restrict__ pe,
                   float* __restrict__ y, int total, int period)
{
    const int i = blockIdx.x * 256 + threadIdx.x;
    if (i < total) y[i] = x[i] + pe[i % period];
}

__global__ __launch_bounds__(256)
void cvt_kernel(const float* __restrict__ x, _Float16* __restrict__ y, int total)
{
    const int i = blockIdx.x * 256 + threadIdx.x;
    if (i < total) y[i] = (_Float16)x[i];
}

// ---------------------------------------------------------------------------
// Host launcher
// ---------------------------------------------------------------------------
extern "C" void kernel_launch(void* const* d_in, const int* in_sizes, int n_in,
                              void* d_out, int out_size, void* d_ws, size_t ws_size,
                              hipStream_t stream)
{
    (void)in_sizes; (void)n_in; (void)out_size; (void)ws_size;

    const float* src      = (const float*)d_in[0];
    const float* tgt      = (const float*)d_in[1];
    const int*   src_mask = (const int*)  d_in[2];   // [B,1,S]
    const int*   tgt_mask = (const int*)  d_in[3];   // [B,S,S]
    const float* pe       = (const float*)d_in[4];   // [1,S,D]
    const float* enc_w    = (const float*)d_in[5];   // [N,4,D,D]
    const float* enc_b    = (const float*)d_in[6];
    const float* enc_ns   = (const float*)d_in[7];
    const float* enc_nb   = (const float*)d_in[8];
    const float* enc_w1   = (const float*)d_in[9];
    const float* enc_b1   = (const float*)d_in[10];
    const float* enc_w2   = (const float*)d_in[11];
    const float* enc_b2   = (const float*)d_in[12];
    const float* dec_w    = (const float*)d_in[13];  // [N,8,D,D]
    const float* dec_b    = (const float*)d_in[14];
    const float* dec_ns   = (const float*)d_in[15];  // [N,3,D]
    const float* dec_nb   = (const float*)d_in[16];
    const float* dec_w1   = (const float*)d_in[17];
    const float* dec_b1   = (const float*)d_in[18];
    const float* dec_w2   = (const float*)d_in[19];
    const float* dec_b2   = (const float*)d_in[20];

    const size_t MSD = (size_t)Bc * Sc * Dc;          // 2M elements
    char* p = (char*)d_ws;
    auto alloc = [&](size_t bytes) {
        void* r = (void*)p; p += (bytes + 255) & ~(size_t)255; return r;
    };
    // fp32 residual-stream buffers
    float* X   = (float*)alloc(MSD * 4);
    float* NM  = (float*)alloc(MSD * 4);
    float* RS  = (float*)alloc(MSD * 4);
    // f16 GEMM-feed buffers
    _Float16* NMh = (_Float16*)alloc(MSD * 2);
    _Float16* Qh  = (_Float16*)alloc(MSD * 2);
    _Float16* Kh  = (_Float16*)alloc(MSD * 2);
    _Float16* Vh  = (_Float16*)alloc(MSD * 2);
    _Float16* Vt  = (_Float16*)alloc(MSD * 2);
    _Float16* AOh = (_Float16*)alloc(MSD * 2);
    _Float16* MEh = (_Float16*)alloc(MSD * 2);
    _Float16* RSh = (_Float16*)alloc(MSD * 2);
    _Float16* FFh = (_Float16*)alloc((size_t)Bc * Sc * DFFc * 2);
    _Float16* SCh = (_Float16*)alloc((size_t)Bc * Hc * Sc * Sc * 2);
    _Float16* WT  = (_Float16*)alloc((size_t)Dc * DFFc * 2);   // weight scratch [N,K]

    const int  MM  = Bc * Sc;                 // 4096 rows
    const dim3 blk(256);
    const dim3 gD (MM / BMT, Dc   / BNT, 1);
    const dim3 gFF(MM / BMT, DFFc / BNT, 1);
    const dim3 gSC(Sc / BMT, Sc / BNT, Bc * Hc);
    const dim3 gPV(Sc / BMT, (DKc + BNT - 1) / BNT, Bc * Hc);
    const int total = Bc * Sc * Dc;

    auto LN = [&](const float* x, const float* s, const float* b) {
        layernorm_kernel<<<MM, blk, 0, stream>>>(x, s, b, NM, NMh);
    };
    auto TCVT = [&](const float* w, int K, int Nn) {
        tcvt_kernel<<<dim3(K / 32, Nn / 32, 1), blk, 0, stream>>>(w, WT, K, Nn);
    };

    // full MHA: outRS = resid + (softmax(QK^T/8, mask) V) Wo + bo
    auto ATTN = [&](const _Float16* aq, const _Float16* akv,
                    const float* Wt4, const float* bt4,
                    const int* mask, long mB, long mQ,
                    const float* resid, float* outRS) {
        const long DD = (long)Dc * Dc;
        // Q/K/V projections (f16 out)
        TCVT(Wt4 + 0 * DD, Dc, Dc);
        gemm_h<false, false, true><<<gD, blk, 0, stream>>>(
            aq, 0, 0, WT, 0, 0, bt4 + 0 * Dc, nullptr, 0, 0,
            Qh, 0, 0, 1, MM, Dc, Dc, Dc, Dc, Dc);
        TCVT(Wt4 + 1 * DD, Dc, Dc);
        gemm_h<false, false, true><<<gD, blk, 0, stream>>>(
            akv, 0, 0, WT, 0, 0, bt4 + 1 * Dc, nullptr, 0, 0,
            Kh, 0, 0, 1, MM, Dc, Dc, Dc, Dc, Dc);
        TCVT(Wt4 + 2 * DD, Dc, Dc);
        gemm_h<false, false, true><<<gD, blk, 0, stream>>>(
            akv, 0, 0, WT, 0, 0, bt4 + 2 * Dc, nullptr, 0, 0,
            Vh, 0, 0, 1, MM, Dc, Dc, Dc, Dc, Dc);
        vtrans_kernel<<<dim3(Sc / 32, DKc / 32, Bc * Hc), blk, 0, stream>>>(Vh, Vt);
        // scores = Q K^T  (A=[S,DK] slice, B=K [S,DK] slice -> [N,K] form)
        gemm_h<false, false, true><<<gSC, blk, 0, stream>>>(
            Qh, (long)Sc * Dc, (long)DKc,
            Kh, (long)Sc * Dc, (long)DKc,
            nullptr, nullptr, 0, 0,
            SCh, (long)Hc * Sc * Sc, (long)Sc * Sc,
            Hc, Sc, Sc, DKc, Dc, Dc, Sc);
        softmax_kernel<<<dim3(Sc, Bc * Hc), blk, 0, stream>>>(
            SCh, mask, mB, mQ, 1.f / 8.f);
        // O = P V   (B = Vt [DK,S], N=64 guarded)
        gemm_h<false, false, true><<<gPV, blk, 0, stream>>>(
            SCh, (long)Hc * Sc * Sc, (long)Sc * Sc,
            Vt, (long)Hc * DKc * Sc, (long)DKc * Sc,
            nullptr, nullptr, 0, 0,
            AOh, (long)Sc * Dc, (long)DKc,
            Hc, Sc, DKc, Sc, Sc, Sc, Dc);
        // out = resid + O Wo + bo  (fp32 out)
        TCVT(Wt4 + 3 * DD, Dc, Dc);
        gemm_h<false, true, false><<<gD, blk, 0, stream>>>(
            AOh, 0, 0, WT, 0, 0, bt4 + 3 * Dc,
            resid, 0, 0, outRS, 0, 0, 1, MM, Dc, Dc, Dc, Dc, Dc);
    };

    // ================= Encoder =================
    add_pe_kernel<<<(total + 255) / 256, blk, 0, stream>>>(src, pe, X, total, Sc * Dc);
    for (int l = 0; l < Nc; ++l) {
        const float* Wl = enc_w  + (size_t)l * 4 * Dc * Dc;
        const float* bl = enc_b  + (size_t)l * 4 * Dc;
        const float* ns = enc_ns + (size_t)l * Dc;
        const float* nb = enc_nb + (size_t)l * Dc;
        LN(X, ns, nb);                                              // xn
        ATTN(NMh, NMh, Wl, bl, src_mask, (long)Sc, 0L, X, RS);      // a
        LN(RS, ns, nb);                                             // an
        TCVT(enc_w1 + (size_t)l * Dc * DFFc, Dc, DFFc);             // w1 [512,2048]
        gemm_h<true, false, true><<<gFF, blk, 0, stream>>>(         // relu -> f16
            NMh, 0, 0, WT, 0, 0, enc_b1 + (size_t)l * DFFc,
            nullptr, 0, 0, FFh, 0, 0, 1, MM, DFFc, Dc, Dc, Dc, DFFc);
        TCVT(enc_w2 + (size_t)l * DFFc * Dc, DFFc, Dc);             // w2 [2048,512]
        gemm_h<false, true, false><<<gD, blk, 0, stream>>>(         // an + ffn
            FFh, 0, 0, WT, 0, 0, enc_b2 + (size_t)l * Dc,
            NM, 0, 0, X, 0, 0, 1, MM, Dc, DFFc, DFFc, DFFc, Dc);
    }
    cvt_kernel<<<(total + 255) / 256, blk, 0, stream>>>(X, MEh, total);  // memory

    // ================= Decoder =================
    add_pe_kernel<<<(total + 255) / 256, blk, 0, stream>>>(tgt, pe, X, total, Sc * Dc);
    for (int l = 0; l < Nc; ++l) {
        const float* Wl = dec_w  + (size_t)l * 8 * Dc * Dc;
        const float* bl = dec_b  + (size_t)l * 8 * Dc;
        const float* ns = dec_ns + (size_t)l * 3 * Dc;
        const float* nb = dec_nb + (size_t)l * 3 * Dc;
        // self-attention (causal)
        LN(X, ns, nb);                                              // xn
        ATTN(NMh, NMh, Wl, bl, tgt_mask, (long)Sc * Sc, (long)Sc, X, RS);  // sa
        // cross-attention (padding mask); residual sa, output in place
        LN(RS, ns + Dc, nb + Dc);                                   // san
        ATTN(NMh, MEh, Wl + 4 * (long)Dc * Dc, bl + 4 * Dc,
             src_mask, (long)Sc, 0L, RS, RS);                       // ca
        // FFN: input = ca (pre-norm), residual = can
        LN(RS, ns + 2 * Dc, nb + 2 * Dc);                           // can
        cvt_kernel<<<(total + 255) / 256, blk, 0, stream>>>(RS, RSh, total);
        TCVT(dec_w1 + (size_t)l * Dc * DFFc, Dc, DFFc);
        gemm_h<true, false, true><<<gFF, blk, 0, stream>>>(
            RSh, 0, 0, WT, 0, 0, dec_b1 + (size_t)l * DFFc,
            nullptr, 0, 0, FFh, 0, 0, 1, MM, DFFc, Dc, Dc, Dc, DFFc);
        TCVT(dec_w2 + (size_t)l * DFFc * Dc, DFFc, Dc);
        gemm_h<false, true, false><<<gD, blk, 0, stream>>>(
            FFh, 0, 0, WT, 0, 0, dec_b2 + (size_t)l * Dc,
            NM, 0, 0, X, 0, 0, 1, MM, Dc, DFFc, DFFc, DFFc, Dc);
    }

    hipMemcpyAsync(d_out, X, MSD * sizeof(float), hipMemcpyDeviceToDevice, stream);
}
